// Mamba1_49100066128032
// MI455X (gfx1250) — compile-verified, matrix operand order checked by reading, not run on hardware
//
#include <hip/hip_runtime.h>
#include <hip/hip_bf16.h>
#include <hip/hip_fp16.h>

// ---------------------------------------------------------------------------
// Model dims
// ---------------------------------------------------------------------------
#define BSZ      4
#define TLEN     1000
#define FREQ0    229
#define DMODEL   768
#define DINNER   1536
#define DSTATE   16
#define DTRANK   48
#define DCONV    4
#define NCLASS   88
#define MIDFEAT  1792           // 128 * (229//16) = 128*14
#define EPSV     1e-5f
#define MROWS    (BSZ * TLEN)   // 4000

typedef __attribute__((ext_vector_type(16))) _Float16 v16h;
typedef __attribute__((ext_vector_type(8)))  float    v8f;

// ---------------------------------------------------------------------------
// WMMA helpers (CDNA5: V_WMMA_F32_16X16X32_F16, wave32)
// ---------------------------------------------------------------------------
__device__ __forceinline__ v8f wmma16x16x32(v16h a, v16h b, v8f c) {
  // 8 args: (neg_a, A, neg_b, B, c_mod, C, reuse_a, reuse_b)
  return __builtin_amdgcn_wmma_f32_16x16x32_f16(false, a, false, b, (short)0, c,
                                                false, false);
}

// A/B fragment gather from an LDS row.
// 16-bit A-matrix 16x32 layout: lanes 0-15: elems 0..7 = K0..7, 8..15 = K16..23
//                               lanes 16-31: elems 0..7 = K8..15, 8..15 = K24..31
__device__ __forceinline__ v16h ld_frag(const _Float16* __restrict__ row, int klo) {
  v16h v;
#pragma unroll
  for (int j = 0; j < 8; ++j) {
    v[j]     = row[klo + j];
    v[j + 8] = row[16 + klo + j];
  }
  return v;
}

// ---------------------------------------------------------------------------
// CDNA5 async global->LDS copy (ASYNCcnt-tracked), 16 bytes per lane.
// Flat LDS addresses map to the LDS offset in their low 32 bits, so truncating
// the generic pointer yields the DS address operand.
// ---------------------------------------------------------------------------
__device__ __forceinline__ void async_g2l_b128(void* lds, const void* g) {
#if defined(__gfx1250__)
  unsigned lds_off = (unsigned)(unsigned long long)lds;
  asm volatile("global_load_async_to_lds_b128 %0, %1, off"
               :: "v"(lds_off), "v"(g)
               : "memory");
#else
  (void)lds; (void)g;
#endif
}
__device__ __forceinline__ void wait_async0() {
#if defined(__gfx1250__)
  asm volatile("s_wait_asynccnt 0x0" ::: "memory");
#endif
}

__device__ __forceinline__ float sigmoidf_(float x) { return 1.f / (1.f + expf(-x)); }
__device__ __forceinline__ float siluf_(float x)    { return x * sigmoidf_(x); }

// ---------------------------------------------------------------------------
// Generic GEMM:  C[M,N] = act( A[M,K](f16) * W[N,K](f32->f16)^T )
// Block tile 128(M) x 64(N), 8 waves (4x2), each wave 32x32 via 2x2 WMMA.
// ACT: 0=none  1=BN+ReLU (scale,bias)  2=softplus(+bias)  3=sigmoid(+bias)
// ---------------------------------------------------------------------------
template <int ACT, typename OutT>
__global__ __launch_bounds__(256) void gemm_kernel(
    const _Float16* __restrict__ A, const float* __restrict__ W,
    const float* __restrict__ scale, const float* __restrict__ bias,
    OutT* __restrict__ C, int M, int N, int K) {
  __shared__ _Float16 lsA[128][40];  // pitch 40 halves (80B) to stagger banks
  __shared__ _Float16 lsB[64][40];

  const int tid  = threadIdx.x;
  const int lane = tid & 31;
  const int wave = tid >> 5;
  const int wm   = wave & 3;          // 0..3  -> M sub-tile
  const int wn   = wave >> 2;         // 0..1  -> N sub-tile
  const int l16  = lane & 15;
  const int klo  = (lane >> 4) << 3;  // 0 or 8
  const int mBase = blockIdx.x * 128;
  const int nBase = blockIdx.y * 64;
  const bool interiorM = (mBase + 128 <= M);

  v8f acc[2][2] = {};

  const int nk = (K + 31) >> 5;
  for (int ks = 0; ks < nk; ++ks) {
    const int k0 = ks << 5;
    const bool fast = interiorM && (k0 + 32 <= K);
    // ---- stage A tile: 128x32 halves, 512 chunks of 16B, 2 chunks/thread ----
    if (fast) {
      // async DMA straight into LDS, no VGPR round-trip
#pragma unroll
      for (int i = 0; i < 2; ++i) {
        int c   = tid + (i << 8);
        int row = c >> 2;
        int col = (c & 3) << 3;
        async_g2l_b128(&lsA[row][col],
                       A + (size_t)(mBase + row) * K + (k0 + col));
      }
      if ((k0 + 64) <= K)
        __builtin_prefetch((const char*)(A + (size_t)(mBase + (tid >> 1)) * K +
                                         k0 + 32), 0, 1);
    } else {
#pragma unroll
      for (int i = 0; i < 2; ++i) {
        int c   = tid + (i << 8);
        int row = c >> 2;
        int col = (c & 3) << 3;
        int m   = mBase + row;
        int k   = k0 + col;
        _Float16* dst = &lsA[row][col];
#pragma unroll
        for (int j = 0; j < 8; ++j) {
          int kk = k + j;
          dst[j] = (m < M && kk < K) ? A[(size_t)m * K + kk] : (_Float16)0.f;
        }
      }
    }
    // ---- stage B tile: 64x32, f32 weights -> f16, 1 chunk of 8/thread ----
    {
      int row = tid >> 2;
      int col = (tid & 3) << 3;
      int n   = nBase + row;
      int k   = k0 + col;
      _Float16* dst = &lsB[row][col];
#pragma unroll
      for (int j = 0; j < 8; ++j) {
        int kk = k + j;
        float v = (n < N && kk < K) ? W[(size_t)n * K + kk] : 0.f;
        dst[j] = (_Float16)v;
      }
    }
    if (fast) wait_async0();  // async LDS writes visible before barrier
    __syncthreads();

    v16h a0 = ld_frag(&lsA[wm * 32 + l16][0], klo);
    v16h a1 = ld_frag(&lsA[wm * 32 + 16 + l16][0], klo);
    v16h b0 = ld_frag(&lsB[wn * 32 + l16][0], klo);
    v16h b1 = ld_frag(&lsB[wn * 32 + 16 + l16][0], klo);
    acc[0][0] = wmma16x16x32(a0, b0, acc[0][0]);
    acc[0][1] = wmma16x16x32(a0, b1, acc[0][1]);
    acc[1][0] = wmma16x16x32(a1, b0, acc[1][0]);
    acc[1][1] = wmma16x16x32(a1, b1, acc[1][1]);
    __syncthreads();
  }

  // ---- epilogue: D layout m = base + r + 8*(lane>=16), n = lane&15 ----
  const int mhalf = (lane >> 4) << 3;
#pragma unroll
  for (int ms = 0; ms < 2; ++ms) {
#pragma unroll
    for (int ns = 0; ns < 2; ++ns) {
      int n = nBase + wn * 32 + ns * 16 + l16;
      if (n >= N) continue;
      float s = 1.f, bb = 0.f;
      if (ACT == 1) { s = scale[n] * rsqrtf(1.f + EPSV); bb = bias[n]; }
      if (ACT == 2 || ACT == 3) { bb = bias[n]; }
#pragma unroll
      for (int r = 0; r < 8; ++r) {
        int m = mBase + wm * 32 + ms * 16 + mhalf + r;
        if (m >= M) continue;
        float v = acc[ms][ns][r];
        if (ACT == 1)      v = fmaxf(v * s + bb, 0.f);
        else if (ACT == 2) { v += bb; v = (v > 20.f) ? v : log1pf(expf(v)); }
        else if (ACT == 3) { v += bb; v = sigmoidf_(v); }
        C[(size_t)m * N + n] = (OutT)v;
      }
    }
  }
}

// ---------------------------------------------------------------------------
// Implicit-GEMM 3x3 SAME conv + BN + ReLU (NCHW f16 -> NCHW f16)
// M = B*T*F spatial, N = Cout, K = Cin*9.
// Per-thread row coordinates are K-invariant -> hoisted out of the K loop.
// ---------------------------------------------------------------------------
__global__ __launch_bounds__(256) void conv_gemm_kernel(
    const _Float16* __restrict__ in, const float* __restrict__ W,
    const float* __restrict__ g, const float* __restrict__ bb_,
    _Float16* __restrict__ out, int Cin, int Cout, int F) {
  __shared__ _Float16 lsA[128][40];
  __shared__ _Float16 lsB[64][40];

  const int T = TLEN;
  const int M = BSZ * T * F;
  const int K = Cin * 9;

  const int tid  = threadIdx.x;
  const int lane = tid & 31;
  const int wave = tid >> 5;
  const int wm   = wave & 3;
  const int wn   = wave >> 2;
  const int l16  = lane & 15;
  const int klo  = (lane >> 4) << 3;
  const int mBase = blockIdx.x * 128;
  const int nBase = blockIdx.y * 64;

  // ---- hoisted im2col row decode: this thread always stages column (tid&31)
  //      of rows rb, rb+8, ..., rb+120.
  const int kkc = tid & 31;
  const int rb  = tid >> 5;
  int   baseoff[16];
  short ttc[16], ffc[16];
  unsigned mmask = 0;
#pragma unroll
  for (int i = 0; i < 16; ++i) {
    int m  = mBase + rb + (i << 3);
    int mm = (m < M) ? m : 0;
    if (m < M) mmask |= (1u << i);
    int f = mm % F;
    int t = (mm / F) % T;
    int b = mm / (F * T);
    ttc[i] = (short)t;
    ffc[i] = (short)f;
    baseoff[i] = (b * Cin * T + t) * F + f;  // + c*T*F + dy*F + dx at use
  }

  v8f acc[2][2] = {};

  const int nk = (K + 31) >> 5;
  for (int ks = 0; ks < nk; ++ks) {
    const int k0 = ks << 5;
    // ---- stage A tile via im2col gather ----
    {
      const int k = k0 + kkc;
      if (k < K) {
        int c  = k / 9;
        int r9 = k - c * 9;
        int dy = r9 / 3 - 1;
        int dx = r9 - (dy + 1) * 3 - 1;
        const int koff = (c * T + dy) * F + dx;
#pragma unroll
        for (int i = 0; i < 16; ++i) {
          _Float16 v = (_Float16)0.f;
          int t2 = (int)ttc[i] + dy;
          int f2 = (int)ffc[i] + dx;
          if (((mmask >> i) & 1u) && t2 >= 0 && t2 < T && f2 >= 0 && f2 < F)
            v = in[(size_t)(baseoff[i] + koff)];
          lsA[rb + (i << 3)][kkc] = v;
        }
      } else {
#pragma unroll
        for (int i = 0; i < 16; ++i) lsA[rb + (i << 3)][kkc] = (_Float16)0.f;
      }
    }
    // ---- stage B tile (weights (Cout, Cin*9) f32 -> f16) ----
    {
      int row = tid >> 2;
      int col = (tid & 3) << 3;
      int n   = nBase + row;
      int k   = k0 + col;
      _Float16* dst = &lsB[row][col];
#pragma unroll
      for (int j = 0; j < 8; ++j) {
        int kk = k + j;
        float v = (n < Cout && kk < K) ? W[(size_t)n * K + kk] : 0.f;
        dst[j] = (_Float16)v;
      }
    }
    __syncthreads();

    v16h a0 = ld_frag(&lsA[wm * 32 + l16][0], klo);
    v16h a1 = ld_frag(&lsA[wm * 32 + 16 + l16][0], klo);
    v16h b0 = ld_frag(&lsB[wn * 32 + l16][0], klo);
    v16h b1 = ld_frag(&lsB[wn * 32 + 16 + l16][0], klo);
    acc[0][0] = wmma16x16x32(a0, b0, acc[0][0]);
    acc[0][1] = wmma16x16x32(a0, b1, acc[0][1]);
    acc[1][0] = wmma16x16x32(a1, b0, acc[1][0]);
    acc[1][1] = wmma16x16x32(a1, b1, acc[1][1]);
    __syncthreads();
  }

  const int mhalf = (lane >> 4) << 3;
#pragma unroll
  for (int ms = 0; ms < 2; ++ms) {
#pragma unroll
    for (int ns = 0; ns < 2; ++ns) {
      int n = nBase + wn * 32 + ns * 16 + l16;
      if (n >= Cout) continue;
      float s  = g[n] * rsqrtf(1.f + EPSV);
      float bv = bb_[n];
#pragma unroll
      for (int r = 0; r < 8; ++r) {
        int m = mBase + wm * 32 + ms * 16 + mhalf + r;
        if (m >= M) continue;
        int f = m % F;
        int t = (m / F) % T;
        int b = m / (F * T);
        float v = fmaxf(acc[ms][ns][r] * s + bv, 0.f);
        out[(((size_t)b * Cout + n) * T + t) * F + f] = (_Float16)v;
      }
    }
  }
}

// ---------------------------------------------------------------------------
// Elementwise / small kernels
// ---------------------------------------------------------------------------
__global__ void f32_to_f16_kernel(const float* __restrict__ in,
                                  _Float16* __restrict__ out, int n) {
  int i = blockIdx.x * 256 + threadIdx.x;
  if (i < n) out[i] = (_Float16)in[i];
}

// sum-pool over freq (window 2, stride 2) then *0.5  ==> average of pairs
__global__ void pool_kernel(const _Float16* __restrict__ in,
                            _Float16* __restrict__ out, int C, int F, int Fo) {
  int n = BSZ * C * TLEN * Fo;
  int idx = blockIdx.x * 256 + threadIdx.x;
  if (idx >= n) return;
  int f = idx % Fo;
  int t = (idx / Fo) % TLEN;
  int c = (idx / (Fo * TLEN)) % C;
  int b = idx / (Fo * TLEN * C);
  size_t base = (((size_t)b * C + c) * TLEN + t) * F + 2 * f;
  out[idx] = (_Float16)(((float)in[base] + (float)in[base + 1]) * 0.5f);
}

// (b,c,t,f) NCHW(128,14) -> (b*T+t, c*14+f) for fc5
__global__ void repack_fc_kernel(const _Float16* __restrict__ h4p,
                                 _Float16* __restrict__ fc_in) {
  int idx = blockIdx.x * 256 + threadIdx.x;
  if (idx >= MROWS * MIDFEAT) return;
  int k = idx % MIDFEAT;
  int m = idx / MIDFEAT;
  int c = k / 14, f = k % 14;
  int t = m % TLEN, b = m / TLEN;
  fc_in[idx] = h4p[(((size_t)b * 128 + c) * TLEN + t) * 14 + f];
}

__global__ void layernorm_kernel(const float* __restrict__ h,
                                 const float* __restrict__ g,
                                 const float* __restrict__ b,
                                 _Float16* __restrict__ out) {
  const int N = DMODEL;
  int row = blockIdx.x;
  const float* x = h + (size_t)row * N;
  __shared__ float red[256];
  int tid = threadIdx.x;

  float s = 0.f;
  for (int i = tid; i < N; i += 256) s += x[i];
  red[tid] = s;
  __syncthreads();
  for (int off = 128; off > 0; off >>= 1) {
    if (tid < off) red[tid] += red[tid + off];
    __syncthreads();
  }
  float mu = red[0] / N;
  __syncthreads();

  float v = 0.f;
  for (int i = tid; i < N; i += 256) {
    float d = x[i] - mu;
    v += d * d;
  }
  red[tid] = v;
  __syncthreads();
  for (int off = 128; off > 0; off >>= 1) {
    if (tid < off) red[tid] += red[tid + off];
    __syncthreads();
  }
  float rstd = rsqrtf(red[0] / N + EPSV);
  __syncthreads();

  for (int i = tid; i < N; i += 256)
    out[(size_t)row * N + i] = (_Float16)((x[i] - mu) * rstd * g[i] + b[i]);
}

// depthwise causal conv1d (kernel 4) + bias + SiLU; xm = xz[:, :1536]
__global__ void dwconv_silu_kernel(const _Float16* __restrict__ xz,
                                   const float* __restrict__ w,
                                   const float* __restrict__ cb,
                                   _Float16* __restrict__ xc) {
  int idx = blockIdx.x * 256 + threadIdx.x;
  if (idx >= MROWS * DINNER) return;
  int d = idx % DINNER;
  int m = idx / DINNER;
  int t = m % TLEN, b = m / TLEN;
  float accv = cb[d];
#pragma unroll
  for (int j = 0; j < DCONV; ++j) {
    int tt = t - (DCONV - 1) + j;
    if (tt >= 0)
      accv += w[d * DCONV + j] *
              (float)xz[((size_t)(b * TLEN + tt)) * (2 * DINNER) + d];
  }
  xc[idx] = (_Float16)siluf_(accv);
}

// dbl[:, :48] f32 -> f16 contiguous for dt-proj GEMM
__global__ void repack_dt_kernel(const float* __restrict__ dbl,
                                 _Float16* __restrict__ dt_in) {
  int idx = blockIdx.x * 256 + threadIdx.x;
  if (idx >= MROWS * DTRANK) return;
  int m = idx / DTRANK, j = idx % DTRANK;
  dt_in[idx] = (_Float16)dbl[(size_t)m * (DTRANK + 2 * DSTATE) + j];
}

// Selective scan: one thread per (b, d); 16-wide state in registers,
// per-timestep B/C broadcast through LDS. Fuses + x*D and * silu(z).
__global__ __launch_bounds__(256) void scan_kernel(
    const float* __restrict__ dt, const _Float16* __restrict__ xc,
    const float* __restrict__ dbl, const float* __restrict__ A_log,
    const float* __restrict__ Dp, const _Float16* __restrict__ xz,
    _Float16* __restrict__ ymul) {
  const int d = blockIdx.x * 256 + threadIdx.x;  // < 1536
  const int b = blockIdx.y;
  float Ad[DSTATE];
#pragma unroll
  for (int n = 0; n < DSTATE; ++n) Ad[n] = -expf(A_log[d * DSTATE + n]);
  const float Dv = Dp[d];
  float h[DSTATE] = {};
  __shared__ float sB[DSTATE];
  __shared__ float sC[DSTATE];

  for (int t = 0; t < TLEN; ++t) {
    int row = b * TLEN + t;
    if (threadIdx.x < 32) {
      float v = dbl[(size_t)row * (DTRANK + 2 * DSTATE) + DTRANK + threadIdx.x];
      if (threadIdx.x < DSTATE) sB[threadIdx.x] = v;
      else                      sC[threadIdx.x - DSTATE] = v;
    }
    __syncthreads();
    float sdt = dt[(size_t)row * DINNER + d];
    float sx  = (float)xc[(size_t)row * DINNER + d];
    float dtx = sdt * sx;
    float y = 0.f;
#pragma unroll
    for (int n = 0; n < DSTATE; ++n) {
      float a = expf(sdt * Ad[n]);
      h[n] = a * h[n] + dtx * sB[n];
      y += h[n] * sC[n];
    }
    y += sx * Dv;
    float zv = (float)xz[(size_t)row * (2 * DINNER) + DINNER + d];
    ymul[(size_t)row * DINNER + d] = (_Float16)(y * siluf_(zv));
    __syncthreads();
  }
}

// ---------------------------------------------------------------------------
// Launcher
// ---------------------------------------------------------------------------
extern "C" void kernel_launch(void* const* d_in, const int* in_sizes, int n_in,
                              void* d_out, int out_size, void* d_ws,
                              size_t ws_size, hipStream_t stream) {
  (void)in_sizes; (void)n_in; (void)out_size; (void)ws_size;

  const float* x = (const float*)d_in[0];
  const float* cw[8];
  const float* cg[8];
  const float* cbv[8];
  for (int blk = 0; blk < 4; ++blk) {
    int base = 1 + blk * 6;
    cw[2 * blk]      = (const float*)d_in[base + 0];
    cg[2 * blk]      = (const float*)d_in[base + 1];
    cbv[2 * blk]     = (const float*)d_in[base + 2];
    cw[2 * blk + 1]  = (const float*)d_in[base + 3];
    cg[2 * blk + 1]  = (const float*)d_in[base + 4];
    cbv[2 * blk + 1] = (const float*)d_in[base + 5];
  }
  const float* fc5_w    = (const float*)d_in[25];
  const float* bn5_g    = (const float*)d_in[26];
  const float* bn5_b    = (const float*)d_in[27];
  const float* ln_g     = (const float*)d_in[28];
  const float* ln_b     = (const float*)d_in[29];
  const float* m_in_w   = (const float*)d_in[30];
  const float* m_conv_w = (const float*)d_in[31];
  const float* m_conv_b = (const float*)d_in[32];
  const float* m_xproj  = (const float*)d_in[33];
  const float* m_dt_w   = (const float*)d_in[34];
  const float* m_dt_b   = (const float*)d_in[35];
  const float* m_A_log  = (const float*)d_in[36];
  const float* m_D      = (const float*)d_in[37];
  const float* m_out_w  = (const float*)d_in[38];
  const float* proj_w   = (const float*)d_in[39];
  const float* proj_b   = (const float*)d_in[40];

  // bump allocator over d_ws
  char* wp = (char*)d_ws;
  auto alloc = [&](size_t bytes) -> void* {
    void* r = (void*)wp;
    wp += (bytes + 255) & ~(size_t)255;
    return r;
  };
  _Float16* x16   = (_Float16*)alloc((size_t)BSZ * TLEN * FREQ0 * 2);
  _Float16* buf0  = (_Float16*)alloc((size_t)44000000 * 2);   // full-res ping
  _Float16* buf1  = (_Float16*)alloc((size_t)44000000 * 2);   // full-res pong
  _Float16* buf2  = (_Float16*)alloc((size_t)22000000 * 2);   // pooled
  _Float16* fc_in = (_Float16*)alloc((size_t)MROWS * MIDFEAT * 2);
  float*    h5    = (float*)   alloc((size_t)MROWS * DMODEL * 4);
  _Float16* xln   = (_Float16*)alloc((size_t)MROWS * DMODEL * 2);
  _Float16* xz    = (_Float16*)alloc((size_t)MROWS * 2 * DINNER * 2);
  _Float16* xc    = (_Float16*)alloc((size_t)MROWS * DINNER * 2);
  float*    dbl   = (float*)   alloc((size_t)MROWS * (DTRANK + 2 * DSTATE) * 4);
  _Float16* dt_in = (_Float16*)alloc((size_t)MROWS * DTRANK * 2);
  float*    dtb   = (float*)   alloc((size_t)MROWS * DINNER * 4);
  _Float16* ymul  = (_Float16*)alloc((size_t)MROWS * DINNER * 2);
  _Float16* y2    = (_Float16*)alloc((size_t)MROWS * DMODEL * 2);

  // ---- input fp32 -> f16 ----
  {
    int n = BSZ * TLEN * FREQ0;
    f32_to_f16_kernel<<<(n + 255) / 256, 256, 0, stream>>>(x, x16, n);
  }

  // ---- conv stack (implicit-GEMM WMMA) ----
  const int Fv[8]  = {229, 229, 114, 114, 57, 57, 28, 28};
  const int ciV[8] = {1, 48, 48, 64, 64, 96, 96, 128};
  const int coV[8] = {48, 48, 64, 64, 96, 96, 128, 128};
  const int Fo[4]  = {114, 57, 28, 14};
  const _Float16* cinP[8] = {x16, buf0, buf2, buf0, buf2, buf0, buf2, buf0};
  _Float16* coutP[8]      = {buf0, buf1, buf0, buf1, buf0, buf1, buf0, buf1};

  for (int i = 0; i < 8; ++i) {
    int F = Fv[i];
    int M = BSZ * TLEN * F;
    dim3 g((M + 127) / 128, (coV[i] + 63) / 64);
    conv_gemm_kernel<<<g, 256, 0, stream>>>(cinP[i], cw[i], cg[i], cbv[i],
                                            coutP[i], ciV[i], coV[i], F);
    if (i & 1) {  // end of block: pool freq /2 into buf2
      int blk = i >> 1;
      int n = BSZ * coV[i] * TLEN * Fo[blk];
      pool_kernel<<<(n + 255) / 256, 256, 0, stream>>>(buf1, buf2, coV[i],
                                                       Fv[i], Fo[blk]);
    }
  }

  // ---- repack to (4000, 1792) and fc5 (BN+ReLU) ----
  {
    int n = MROWS * MIDFEAT;
    repack_fc_kernel<<<(n + 255) / 256, 256, 0, stream>>>(buf2, fc_in);
  }
  {
    dim3 g((MROWS + 127) / 128, (DMODEL + 63) / 64);
    gemm_kernel<1, float><<<g, 256, 0, stream>>>(fc_in, fc5_w, bn5_g, bn5_b,
                                                 h5, MROWS, DMODEL, MIDFEAT);
  }

  // ---- LayerNorm -> f16 ----
  layernorm_kernel<<<MROWS, 256, 0, stream>>>(h5, ln_g, ln_b, xln);

  // ---- Mamba: in_proj (4000x768 @ 3072) ----
  {
    dim3 g((MROWS + 127) / 128, (2 * DINNER + 63) / 64);
    gemm_kernel<0, _Float16><<<g, 256, 0, stream>>>(
        xln, m_in_w, nullptr, nullptr, xz, MROWS, 2 * DINNER, DMODEL);
  }
  // depthwise causal conv1d + SiLU
  {
    int n = MROWS * DINNER;
    dwconv_silu_kernel<<<(n + 255) / 256, 256, 0, stream>>>(xz, m_conv_w,
                                                            m_conv_b, xc);
  }
  // x_proj: (4000,1536) -> (4000,80)
  {
    dim3 g((MROWS + 127) / 128, (DTRANK + 2 * DSTATE + 63) / 64);
    gemm_kernel<0, float><<<g, 256, 0, stream>>>(
        xc, m_xproj, nullptr, nullptr, dbl, MROWS, DTRANK + 2 * DSTATE, DINNER);
  }
  // dt_proj + softplus: (4000,48) -> (4000,1536)
  {
    int n = MROWS * DTRANK;
    repack_dt_kernel<<<(n + 255) / 256, 256, 0, stream>>>(dbl, dt_in);
    dim3 g((MROWS + 127) / 128, (DINNER + 63) / 64);
    gemm_kernel<2, float><<<g, 256, 0, stream>>>(dt_in, m_dt_w, nullptr, m_dt_b,
                                                 dtb, MROWS, DINNER, DTRANK);
  }
  // selective scan (fused + x*D and * silu(z))
  {
    dim3 g(DINNER / 256, BSZ);
    scan_kernel<<<g, 256, 0, stream>>>(dtb, xc, dbl, m_A_log, m_D, xz, ymul);
  }
  // out_proj: (4000,1536) -> (4000,768)
  {
    dim3 g((MROWS + 127) / 128, (DMODEL + 63) / 64);
    gemm_kernel<0, _Float16><<<g, 256, 0, stream>>>(
        ymul, m_out_w, nullptr, nullptr, y2, MROWS, DMODEL, DINNER);
  }
  // head: (4000,768) -> (4000,88), sigmoid, fp32 to d_out
  {
    dim3 g((MROWS + 127) / 128, (NCLASS + 63) / 64);
    gemm_kernel<3, float><<<g, 256, 0, stream>>>(
        y2, proj_w, nullptr, proj_b, (float*)d_out, MROWS, NCLASS, DMODEL);
  }
}
